// LISTERDecoder_49091476193437
// MI455X (gfx1250) — matrix operand order repaired and checked by reading, not compiled, not run on hardware
//
#include <hip/hip_runtime.h>
#include <math.h>

// ---------------- CDNA5 WMMA types ----------------
typedef __attribute__((ext_vector_type(16))) __bf16 v16bf;
typedef __attribute__((ext_vector_type(8)))  float  v8f;

#define C_DIM 512
#define NHEAD 8
#define HDIM  64
#define WIN   11
#define PADW  5
#define NCLS  97
#define BB    128
#define TT    32
#define NTOK  256
#define NEXT  257

__device__ __forceinline__ unsigned short f2bf(float f) {
  unsigned int u = __float_as_uint(f);
  u += 0x7FFFu + ((u >> 16) & 1u);   // round-to-nearest-even
  return (unsigned short)(u >> 16);
}

// ---------------------------------------------------------------------------
// Generic batched GEMM:  C[M,N] = A[M,K] * op(B) + bias[N] + *sbias
// op(B): transB=0 -> B is (K,N) row-major ; transB=1 -> B is (N,K) row-major
// Tile 128x64x32, 256 threads = 8 waves, each wave one 16-row strip,
// 4 x v_wmma_f32_16x16x32_bf16 per K-step.
// Fragment layouts per CDNA5 ISA 7.12.2:
//   A(16x32 bf16): lane = m + 16*((k>>3)&1), elem = (k&7)|((k&16)>>1)
//   B(32x16 bf16): lane = n + 16*((k>>3)&1), elem = (k&7)|((k&16)>>1)
//   C(16x16 f32):  vgpr g: m = g + 8*(lane>>4), n = lane&15
// ---------------------------------------------------------------------------
__global__ __launch_bounds__(256) void gemm_kernel(
    const float* __restrict__ A, int lda, long long sA,
    const float* __restrict__ B, int ldb, long long sB, int transB,
    float* __restrict__ Cc, int ldc, long long sC,
    const float* __restrict__ bias, const float* __restrict__ sbias,
    int M, int N, int K)
{
  __shared__ __align__(16) unsigned short Al[8][32][16];
  __shared__ __align__(16) unsigned short Bl[4][32][16];
  const int tid  = threadIdx.x;
  const int wave = tid >> 5;
  const int lane = tid & 31;
  const int bm0 = blockIdx.y * 128;
  const int bn0 = blockIdx.x * 64;
  const float* Ab = A + (long long)blockIdx.z * sA;
  const float* Bb = B + (long long)blockIdx.z * sB;
  float*       Cb = Cc + (long long)blockIdx.z * sC;

  v8f acc[4];
#pragma unroll
  for (int i = 0; i < 4; ++i)
#pragma unroll
    for (int j = 0; j < 8; ++j) acc[i][j] = 0.f;

  for (int k0 = 0; k0 < K; k0 += 32) {
    // stage A tile 128x32 (f32 -> bf16, fragment-major in LDS)
    for (int i = tid; i < 128 * 32; i += 256) {
      int m = i >> 5, k = i & 31;
      int gm = bm0 + m, gk = k0 + k;
      float v = (gm < M && gk < K) ? Ab[(long long)gm * lda + gk] : 0.f;
      Al[m >> 4][(m & 15) + (((k >> 3) & 1) << 4)][(k & 7) | ((k & 16) >> 1)] = f2bf(v);
    }
    // stage B tile 32x64
    for (int i = tid; i < 32 * 64; i += 256) {
      int k = i >> 6, n = i & 63;
      int gn = bn0 + n, gk = k0 + k;
      float v = 0.f;
      if (gn < N && gk < K)
        v = transB ? Bb[(long long)gn * ldb + gk] : Bb[(long long)gk * ldb + gn];
      Bl[n >> 4][(n & 15) + (((k >> 3) & 1) << 4)][(k & 7) | ((k & 16) >> 1)] = f2bf(v);
    }
    __syncthreads();
    v16bf a = *reinterpret_cast<const v16bf*>(&Al[wave][lane][0]);
#pragma unroll
    for (int nt = 0; nt < 4; ++nt) {
      v16bf b = *reinterpret_cast<const v16bf*>(&Bl[nt][lane][0]);
      acc[nt] = __builtin_amdgcn_wmma_f32_16x16x32_bf16(
          false, a, false, b, (short)0, acc[nt], false, false);
    }
    __syncthreads();
  }

  float sbv = sbias ? *sbias : 0.f;
#pragma unroll
  for (int nt = 0; nt < 4; ++nt) {
    int gn = bn0 + nt * 16 + (lane & 15);
    if (gn >= N) continue;
    float bv = (bias ? bias[gn] : 0.f) + sbv;
#pragma unroll
    for (int g = 0; g < 8; ++g) {
      int gm = bm0 + wave * 16 + g + ((lane >> 4) << 3);
      if (gm < M) Cb[(long long)gm * ldc + gn] = acc[nt][g] + bv;
    }
  }
}

// ---------------- x_ext[b,n,c] = x[b,c,n] (n<256) | eos[c] (n==256) --------
__global__ void build_xext(const float* __restrict__ x,
                           const float* __restrict__ eos,
                           float* __restrict__ xe)
{
  long long idx = (long long)blockIdx.x * 256 + threadIdx.x;
  if (idx >= (long long)BB * NEXT * C_DIM) return;
  int c = (int)(idx & 511);
  long long r = idx >> 9;
  int n = (int)(r % NEXT);
  int b = (int)(r / NEXT);
  xe[idx] = (n < NTOK) ? x[((long long)b * C_DIM + c) * NTOK + n] : eos[c];
}

// ---------------- g[b,c] = mean_n x_ext[b,n,c] -----------------------------
__global__ void mean_kernel(const float* __restrict__ xe, float* __restrict__ g)
{
  int idx = blockIdx.x * 256 + threadIdx.x;   // over BB*C
  int b = idx >> 9, c = idx & 511;
  const float* p = xe + (long long)b * NEXT * C_DIM + c;
  float s = 0.f;
  for (int n = 0; n < NTOK; ++n) s += p[(long long)n * C_DIM];
  g[idx] = s * (1.f / 256.f);
}

// ---------------- start logits: dot(k_feat[b,n,:], q_start[b,:])/sqrt(C) ---
__global__ __launch_bounds__(256) void start_logits(
    const float* __restrict__ kfeat, const float* __restrict__ qstart,
    float* __restrict__ smap)
{
  __shared__ float qs[C_DIM];
  int b = blockIdx.x, tid = threadIdx.x;
  qs[tid]       = qstart[b * C_DIM + tid];
  qs[tid + 256] = qstart[b * C_DIM + tid + 256];
  __syncthreads();
  for (int n = tid; n < NEXT; n += 256) {
    const float* kp = kfeat + ((long long)b * NEXT + n) * C_DIM;
    float s = 0.f;
    for (int c = 0; c < C_DIM; ++c) s += kp[c] * qs[c];
    smap[b * NEXT + n] = s * 0.04419417382415922f;  // 1/sqrt(512)
  }
}

// ---------------- row softmax (len elements, pre-scale) --------------------
__global__ __launch_bounds__(256) void row_softmax(float* __restrict__ x,
                                                   int len, float scale)
{
  __shared__ float red[256];
  __shared__ float sh_m, sh_s;
  float* p = x + (long long)blockIdx.x * len;
  int tid = threadIdx.x;
  float lmax = -3.4e38f;
  for (int i = tid; i < len; i += 256) lmax = fmaxf(lmax, p[i] * scale);
  red[tid] = lmax; __syncthreads();
  for (int s = 128; s > 0; s >>= 1) { if (tid < s) red[tid] = fmaxf(red[tid], red[tid + s]); __syncthreads(); }
  if (tid == 0) sh_m = red[0];
  __syncthreads();
  float m = sh_m, lsum = 0.f;
  for (int i = tid; i < len; i += 256) { float e = expf(p[i] * scale - m); p[i] = e; lsum += e; }
  red[tid] = lsum; __syncthreads();
  for (int s = 128; s > 0; s >>= 1) { if (tid < s) red[tid] += red[tid + s]; __syncthreads(); }
  if (tid == 0) sh_s = red[0];
  __syncthreads();
  float inv = 1.f / sh_s;
  for (int i = tid; i < len; i += 256) p[i] *= inv;
}

// ---------------- alignment scan: T sequential vec-mat steps ---------------
// nb_map row 256 is one-hot at col 256, handled analytically.
__global__ __launch_bounds__(288) void align_scan(
    const float* __restrict__ smap, const float* __restrict__ nb,
    float* __restrict__ maps, float* __restrict__ masks)
{
  __shared__ float cm[NEXT];
  __shared__ int fin;
  int b = blockIdx.x, tid = threadIdx.x;
  if (tid < NEXT) cm[tid] = smap[b * NEXT + tid];
  if (tid == 0) fin = 0;
  __syncthreads();
  for (int t = 0; t < TT; ++t) {
    if (tid < NEXT) maps[((long long)(b * TT + t)) * NEXT + tid] = cm[tid];
    if (tid == 0) masks[b * TT + t] = (fin == 0) ? 1.f : 0.f;
    float last = cm[256];
    float nv = 0.f;
    if (tid < NEXT) {
      const float* nbb = nb + (long long)b * NTOK * NEXT;
      for (int n = 0; n < NTOK; ++n) nv += cm[n] * nbb[(long long)n * NEXT + tid];
      if (tid == 256) nv += last;
    }
    __syncthreads();
    if (tid == 0 && last > 0.6f) fin += 1;
    if (tid < NEXT) cm[tid] = nv;
    __syncthreads();
  }
}

// ---------------- LayerNorm over C=512 (optional mask multiply) ------------
__global__ __launch_bounds__(256) void layernorm_k(
    const float* __restrict__ x, const float* __restrict__ gamma,
    const float* __restrict__ beta, const float* __restrict__ masks,
    float* __restrict__ y)
{
  __shared__ float red[256];
  __shared__ float sh_mean, sh_inv;
  int row = blockIdx.x, tid = threadIdx.x;
  const float* p = x + (long long)row * C_DIM;
  float a = p[tid], b2 = p[tid + 256];
  red[tid] = a + b2; __syncthreads();
  for (int s = 128; s > 0; s >>= 1) { if (tid < s) red[tid] += red[tid + s]; __syncthreads(); }
  if (tid == 0) sh_mean = red[0] * (1.f / 512.f);
  __syncthreads();
  float m = sh_mean;
  float d1 = a - m, d2 = b2 - m;
  red[tid] = d1 * d1 + d2 * d2; __syncthreads();
  for (int s = 128; s > 0; s >>= 1) { if (tid < s) red[tid] += red[tid + s]; __syncthreads(); }
  if (tid == 0) sh_inv = rsqrtf(red[0] * (1.f / 512.f) + 1e-5f);
  __syncthreads();
  float inv = sh_inv;
  float mk = masks ? masks[row] : 1.f;
  y[(long long)row * C_DIM + tid]       = (d1 * inv * gamma[tid] + beta[tid]) * mk;
  y[(long long)row * C_DIM + tid + 256] = (d2 * inv * gamma[tid + 256] + beta[tid + 256]) * mk;
}

// ---------------- local windowed attention core ----------------------------
// k[b,t,w] = kvb_nobias[b, t+w-5] (0 if OOB) + poskv[w] + kv_b   (v likewise)
__global__ __launch_bounds__(512) void local_attn(
    const float* __restrict__ q, const float* __restrict__ kvb,
    const float* __restrict__ poskv, const float* __restrict__ kv_bias,
    float* __restrict__ out)
{
  __shared__ float qs[C_DIM];
  __shared__ float ks[WIN][C_DIM];
  __shared__ float vs[WIN][C_DIM];
  __shared__ float al[NHEAD][WIN + 1];
  int bt = blockIdx.x;
  int b = bt >> 5, t = bt & 31;
  int tid = threadIdx.x;
  qs[tid] = q[(long long)bt * C_DIM + tid];
  for (int i = tid; i < WIN * C_DIM; i += 512) {
    int w = i >> 9, c = i & 511;
    int src = t + w - PADW;
    float kx = 0.f, vx = 0.f;
    if (src >= 0 && src < TT) {
      const float* kp = kvb + ((long long)(b * TT + src)) * (2 * C_DIM);
      kx = kp[c];
      vx = kp[C_DIM + c];
    }
    if (poskv) { kx += poskv[w * 2 * C_DIM + c]; vx += poskv[w * 2 * C_DIM + C_DIM + c]; }
    kx += kv_bias[c];
    vx += kv_bias[C_DIM + c];
    ks[w][c] = kx;
    vs[w][c] = vx;
  }
  __syncthreads();
  if (tid < NHEAD * WIN) {
    int h = tid / WIN, w = tid % WIN;
    float s = 0.f;
    for (int d = 0; d < HDIM; ++d) s += qs[h * HDIM + d] * ks[w][h * HDIM + d];
    al[h][w] = s * 0.037467113637474544f;  // ln(11)/64
  }
  __syncthreads();
  if (tid < NHEAD) {
    float mx = -3.4e38f;
    for (int w = 0; w < WIN; ++w) mx = fmaxf(mx, al[tid][w]);
    float sum = 0.f;
    for (int w = 0; w < WIN; ++w) { float e = expf(al[tid][w] - mx); al[tid][w] = e; sum += e; }
    float inv = 1.f / sum;
    for (int w = 0; w < WIN; ++w) al[tid][w] *= inv;
  }
  __syncthreads();
  int h = tid >> 6, d = tid & 63;
  float o = 0.f;
  for (int w = 0; w < WIN; ++w) o += al[h][w] * vs[w][h * HDIM + d];
  out[(long long)bt * C_DIM + tid] = o;
}

// ---------------- elementwise helpers --------------------------------------
__global__ void maskmul_k(float* __restrict__ f, const float* __restrict__ masks) {
  long long idx = (long long)blockIdx.x * 256 + threadIdx.x;
  f[idx] *= masks[(int)(idx >> 9)];
}
__global__ void resid1_k(float* __restrict__ f, const float* __restrict__ y,
                         const float* __restrict__ g1, const float* __restrict__ masks) {
  long long idx = (long long)blockIdx.x * 256 + threadIdx.x;
  int row = (int)(idx >> 9), c = (int)(idx & 511);
  f[idx] += g1[c] * y[idx] * masks[row];
}
__global__ void resid2_k(float* __restrict__ f, const float* __restrict__ m_in,
                         const float* __restrict__ g2, const float* __restrict__ masks) {
  long long idx = (long long)blockIdx.x * 256 + threadIdx.x;
  int row = (int)(idx >> 9), c = (int)(idx & 511);
  f[idx] = (f[idx] + g2[c] * m_in[idx]) * masks[row];
}
__global__ void gelu_k(float* __restrict__ h, long long n) {
  long long idx = (long long)blockIdx.x * 256 + threadIdx.x;
  if (idx < n) {
    float v = h[idx];
    h[idx] = 0.5f * v * (1.f + erff(v * 0.7071067811865475f));
  }
}

// ---------------------------------------------------------------------------
struct BlockP {
  const float *ln1_g, *ln1_b, *q_w, *q_b, *kv_w, *kv_b, *proj_w, *proj_b,
              *ln2_g, *ln2_b, *fc1_w, *fc1_b, *fc2_w, *fc2_b, *g1, *g2, *pos_bias;
};

static inline void gemm(hipStream_t s,
                        const float* A, int lda, long long sA,
                        const float* B, int ldb, long long sB, int transB,
                        float* Cc, int ldc, long long sC,
                        const float* bias, const float* sbias,
                        int M, int N, int K, int batch)
{
  dim3 grid((N + 63) / 64, (M + 127) / 128, batch);
  gemm_kernel<<<grid, dim3(256), 0, s>>>(A, lda, sA, B, ldb, sB, transB,
                                         Cc, ldc, sC, bias, sbias, M, N, K);
}

extern "C" void kernel_launch(void* const* d_in, const int* in_sizes, int n_in,
                              void* d_out, int out_size, void* d_ws, size_t ws_size,
                              hipStream_t stream)
{
  (void)in_sizes; (void)n_in; (void)out_size; (void)ws_size;
  int ii = 0;
  auto F = [&]() { return (const float*)d_in[ii++]; };
  const float* x    = F();     // (B,C,H,W)
  const float* eos  = F();     // (C)
  const float* qw   = F();     // (C,C)
  const float* qb   = F();
  const float* kw   = F();
  const float* kb   = F();
  const float* bilw = F();
  const float* bilb = F();     // scalar
  const float* clsw = F();
  const float* clsb = F();
  BlockP bp[2];
  for (int l = 0; l < 2; ++l) {
    bp[l].ln1_g = F(); bp[l].ln1_b = F();
    bp[l].q_w   = F(); bp[l].q_b   = F();
    bp[l].kv_w  = F(); bp[l].kv_b  = F();
    bp[l].proj_w= F(); bp[l].proj_b= F();
    bp[l].ln2_g = F(); bp[l].ln2_b = F();
    bp[l].fc1_w = F(); bp[l].fc1_b = F();
    bp[l].fc2_w = F(); bp[l].fc2_b = F();
    bp[l].g1    = F(); bp[l].g2    = F();
    bp[l].pos_bias = (l == 0) ? F() : nullptr;
  }
  // d_in[ii] = max_char (int, ==32) — compile-time constant TT used.

  // ---------------- workspace layout (floats) ----------------
  float* W = (float*)d_ws;
  size_t off = 0;
  auto take = [&](size_t n) { size_t r = off; off += (n + 63) & ~(size_t)63; return W + r; };
  const size_t SZ_XEXT = (size_t)BB * NEXT * C_DIM;   // 16.8M f
  const size_t SZ_SEQ  = (size_t)BB * NTOK * C_DIM;   // 16.8M f
  float* x_ext  = take(SZ_XEXT);
  float* kfeat  = take(SZ_XEXT);
  float* qfeat  = take(SZ_SEQ);
  float* tmpq   = take(SZ_SEQ);
  float* nb     = take((size_t)BB * NTOK * NEXT);
  float* gmean  = take((size_t)BB * C_DIM);
  float* qstart = take((size_t)BB * C_DIM);
  float* smap   = take((size_t)BB * NEXT);
  float* cmaps  = take((size_t)BB * TT * NEXT);
  float* masks  = take((size_t)BB * TT);
  float* poskv  = take((size_t)WIN * 2 * C_DIM);
  // reuse dead regions for block-phase tensors
  const size_t SEQROW = (size_t)BB * TT * C_DIM;      // 2,097,152
  float* feats = qfeat;                // qfeat dead after nb
  float* ybuf  = qfeat + 1 * SEQROW;
  float* qbuf  = qfeat + 2 * SEQROW;
  float* abuf  = qfeat + 3 * SEQROW;
  float* mbuf  = qfeat + 4 * SEQROW;
  float* kvb   = tmpq;                 // tmpq dead after nb
  float* hbuf  = tmpq + 2 * SEQROW;    // 4 * SEQROW floats

  // ---------------- pipeline ----------------
  {
    long long total = (long long)BB * NEXT * C_DIM;
    build_xext<<<dim3((unsigned)((total + 255) / 256)), 256, 0, stream>>>(x, eos, x_ext);
  }
  mean_kernel<<<dim3(BB * C_DIM / 256), 256, 0, stream>>>(x_ext, gmean);

  // k_feat = x_ext @ k_w^T + k_b
  gemm(stream, x_ext, C_DIM, 0, kw, C_DIM, 0, 1, kfeat, C_DIM, 0,
       kb, nullptr, BB * NEXT, C_DIM, C_DIM, 1);
  // q_start = g @ q_w^T + q_b
  gemm(stream, gmean, C_DIM, 0, qw, C_DIM, 0, 1, qstart, C_DIM, 0,
       qb, nullptr, BB, C_DIM, C_DIM, 1);
  // q_feat = xf @ q_w^T + q_b   (batched: rows of x_ext minus eos row)
  gemm(stream, x_ext, C_DIM, (long long)NEXT * C_DIM, qw, C_DIM, 0, 1,
       qfeat, C_DIM, (long long)NTOK * C_DIM, qb, nullptr, NTOK, C_DIM, C_DIM, BB);

  start_logits<<<dim3(BB), 256, 0, stream>>>(kfeat, qstart, smap);
  row_softmax<<<dim3(BB), 256, 0, stream>>>(smap, NEXT, 1.f);

  // tmp = q_feat @ bil_w
  gemm(stream, qfeat, C_DIM, 0, bilw, C_DIM, 0, 0, tmpq, C_DIM, 0,
       nullptr, nullptr, BB * NTOK, C_DIM, C_DIM, 1);
  // nb = tmp @ k_feat^T + bil_b   (batched)
  gemm(stream, tmpq, C_DIM, (long long)NTOK * C_DIM, kfeat, C_DIM,
       (long long)NEXT * C_DIM, 1, nb, NEXT, (long long)NTOK * NEXT,
       nullptr, bilb, NTOK, NEXT, C_DIM, BB);
  row_softmax<<<dim3(BB * NTOK), 256, 0, stream>>>(nb, NEXT, 0.04419417382415922f);

  align_scan<<<dim3(BB), 288, 0, stream>>>(smap, nb, cmaps, masks);

  // char_feats = char_maps @ x_ext  (batched), then * masks
  gemm(stream, cmaps, NEXT, (long long)TT * NEXT, x_ext, C_DIM,
       (long long)NEXT * C_DIM, 0, feats, C_DIM, (long long)TT * C_DIM,
       nullptr, nullptr, TT, C_DIM, NEXT, BB);
  maskmul_k<<<dim3((unsigned)(SEQROW / 256)), 256, 0, stream>>>(feats, masks);

  const int ROWS = BB * TT;  // 4096
  for (int l = 0; l < 2; ++l) {
    const BlockP& p = bp[l];
    // y2 = LN(feats) * mask
    layernorm_k<<<dim3(ROWS), 256, 0, stream>>>(feats, p.ln1_g, p.ln1_b, masks, ybuf);
    // q = y2 @ q_w^T + q_b
    gemm(stream, ybuf, C_DIM, 0, p.q_w, C_DIM, 0, 1, qbuf, C_DIM, 0,
         p.q_b, nullptr, ROWS, C_DIM, C_DIM, 1);
    // kv (no bias) = y2 @ kv_w^T
    gemm(stream, ybuf, C_DIM, 0, p.kv_w, C_DIM, 0, 1, kvb, 2 * C_DIM, 0,
         nullptr, nullptr, ROWS, 2 * C_DIM, C_DIM, 1);
    if (p.pos_bias) {
      // poskv (no bias) = pos_bias @ kv_w^T  (11 x 1024)
      gemm(stream, p.pos_bias, C_DIM, 0, p.kv_w, C_DIM, 0, 1, poskv, 2 * C_DIM, 0,
           nullptr, nullptr, WIN, 2 * C_DIM, C_DIM, 1);
    }
    local_attn<<<dim3(ROWS), 512, 0, stream>>>(qbuf, kvb,
        p.pos_bias ? poskv : nullptr, p.kv_b, abuf);
    // proj
    gemm(stream, abuf, C_DIM, 0, p.proj_w, C_DIM, 0, 1, ybuf, C_DIM, 0,
         p.proj_b, nullptr, ROWS, C_DIM, C_DIM, 1);
    resid1_k<<<dim3((unsigned)(SEQROW / 256)), 256, 0, stream>>>(feats, ybuf, p.g1, masks);
    // MLP
    layernorm_k<<<dim3(ROWS), 256, 0, stream>>>(feats, p.ln2_g, p.ln2_b, nullptr, ybuf);
    gemm(stream, ybuf, C_DIM, 0, p.fc1_w, C_DIM, 0, 1, hbuf, 4 * C_DIM, 0,
         p.fc1_b, nullptr, ROWS, 4 * C_DIM, C_DIM, 1);
    {
      long long nh = (long long)ROWS * 4 * C_DIM;
      gelu_k<<<dim3((unsigned)((nh + 255) / 256)), 256, 0, stream>>>(hbuf, nh);
    }
    gemm(stream, hbuf, 4 * C_DIM, 0, p.fc2_w, 4 * C_DIM, 0, 1, mbuf, C_DIM, 0,
         p.fc2_b, nullptr, ROWS, C_DIM, 4 * C_DIM, 1);
    resid2_k<<<dim3((unsigned)(SEQROW / 256)), 256, 0, stream>>>(feats, mbuf, p.g2, masks);
  }

  // logits = feats @ cls_w^T + cls_b
  gemm(stream, feats, C_DIM, 0, clsw, C_DIM, 0, 1, (float*)d_out, NCLS, 0,
       clsb, nullptr, ROWS, NCLS, C_DIM, 1);
}
